// RGCNLayer_29403346108690
// MI455X (gfx1250) — matrix operand order repaired and checked by reading, not compiled, not run on hardware
//
#include <hip/hip_runtime.h>
#include <hip/hip_bf16.h>
#include <math.h>

typedef __attribute__((ext_vector_type(16))) __bf16 v16bf;
typedef __attribute__((ext_vector_type(8)))  float  v8f;

#define NNODES 50000
#define NEDGES 500000
#define NREL   8
#define FIN    128
#define FOUT   128
#define NGLOB  256

// ---------- helpers ----------

static __device__ __forceinline__ unsigned short f2bf(float f) {
    unsigned int u = __float_as_uint(f);
    unsigned int r = u + 0x7FFFu + ((u >> 16) & 1u);   // round-to-nearest-even
    return (unsigned short)(r >> 16);
}

static __device__ __forceinline__ void atomicMaxFloat(float* addr, float val) {
    if (val >= 0.0f)
        atomicMax((int*)addr, __float_as_int(val));
    else
        atomicMin((unsigned int*)addr, __float_as_uint(val));
}

union Frag8 { unsigned int u[8]; v16bf v; };

// One 16x16 (f32 acc) tile, K-loop in steps of 32, bf16 A/B.
// A: row-major [16 x K] (per-lane row pointer Arow = base + M*K, M = lane&15)
//    lane<16 holds K {0..15}+k0 ; lane>=16 holds K {16..31}+k0
// B: stored transposed as [O x K] row-major (per-lane row pointer Brow = base + N*K)
//    lane<16 holds K {0..15}+k0 ; lane>=16 holds K {16..31}+k0
static __device__ __forceinline__ void wmma_k_loop(
    const unsigned short* __restrict__ Arow,
    const unsigned short* __restrict__ Brow,
    int K, int half, v8f& acc)
{
    for (int k0 = 0; k0 < K; k0 += 32) {
        Frag8 a, b;
        const unsigned int* pa = (const unsigned int*)(Arow + k0 + half * 16);
        const unsigned int* qa = (const unsigned int*)(Arow + k0 + half * 16 + 8);
#pragma unroll
        for (int i = 0; i < 4; ++i) { a.u[i] = pa[i]; a.u[4 + i] = qa[i]; }
        const unsigned int* pb = (const unsigned int*)(Brow + k0 + half * 16);
#pragma unroll
        for (int i = 0; i < 8; ++i) b.u[i] = pb[i];
        acc = __builtin_amdgcn_wmma_f32_16x16x32_bf16(
            false, a.v, false, b.v, (short)0, acc, false, false);
    }
}

// ---------- kernels ----------

// Convert h/weights to bf16 (weights transposed to [O][K]); init accumulators.
__global__ void prep_kernel(const float* __restrict__ h,
                            const float* __restrict__ weight,
                            const float* __restrict__ loopw,
                            const float* __restrict__ Wn,
                            unsigned short* __restrict__ hB,
                            unsigned short* __restrict__ wT,
                            unsigned short* __restrict__ loopT,
                            unsigned short* __restrict__ WnT,
                            float* __restrict__ segmax,
                            float* __restrict__ denom,
                            float* __restrict__ agg,
                            float* __restrict__ sum_nodes)
{
    long long i = (long long)blockIdx.x * blockDim.x + threadIdx.x;
    const long long TOT = (long long)NNODES * FIN;   // 6.4M, covers everything below
    if (i >= TOT) return;
    hB[i]  = f2bf(h[i]);
    agg[i] = 0.0f;                                   // NNODES*FOUT == NNODES*FIN
    if (i < (long long)NREL * FIN * FOUT) {          // weight[r][k][o] -> wT[r][o][k]
        int r = (int)(i / (FIN * FOUT));
        int rem = (int)(i % (FIN * FOUT));
        int k = rem / FOUT, o = rem % FOUT;
        wT[((long long)r * FOUT + o) * FIN + k] = f2bf(weight[i]);
    }
    if (i < FIN * FOUT) {                            // [k][o] -> [o][k]
        int k = (int)i / FOUT, o = (int)i % FOUT;
        loopT[o * FIN + k] = f2bf(loopw[i]);
        WnT[o * FIN + k]   = f2bf(Wn[i]);
    }
    if (i < NNODES) { segmax[i] = -INFINITY; denom[i] = 0.0f; }
    if (i < FOUT)   sum_nodes[i] = 0.0f;
}

// Ht[n][r][o] = h[n][:] @ weight[r][:][o]   -- grid (3125, R), 8 waves = 8 col tiles
__global__ void ht_gemm_kernel(const unsigned short* __restrict__ hB,
                               const unsigned short* __restrict__ wT,
                               float* __restrict__ Ht)
{
    const int lane = threadIdx.x & 31;
    const int wave = threadIdx.x >> 5;       // tile_n: 0..7
    const int half = lane >> 4;
    const int mr   = lane & 15;
    const int tile_m = blockIdx.x;
    const int r      = blockIdx.y;

    const unsigned short* Arow = hB + (long long)(tile_m * 16 + mr) * FIN;
    const unsigned short* Brow = wT + ((long long)r * FOUT + wave * 16 + mr) * FIN;

    v8f acc = {};
    wmma_k_loop(Arow, Brow, FIN, half, acc);

    const int col = wave * 16 + mr;
#pragma unroll
    for (int v = 0; v < 8; ++v) {
        int row = tile_m * 16 + v + half * 8;
        Ht[((long long)row * NREL + r) * FOUT + col] = acc[v];
    }
}

// s1[n,r] = Ht[n,r,:]·Wa[0:128] ; s2[n,r] = Ht[n,r,:]·Wa[128:256]. One wave per (n,r).
__global__ void score_kernel(const float* __restrict__ Ht,
                             const float* __restrict__ Wa,
                             float* __restrict__ s1,
                             float* __restrict__ s2)
{
    const int wid  = (int)(((long long)blockIdx.x * blockDim.x + threadIdx.x) >> 5);
    const int lane = threadIdx.x & 31;
    if (wid >= NNODES * NREL) return;
    float4 v  = ((const float4*)(Ht + (long long)wid * FOUT))[lane];
    float4 w1 = ((const float4*)Wa)[lane];
    float4 w2 = ((const float4*)(Wa + FOUT))[lane];
    float a = v.x * w1.x + v.y * w1.y + v.z * w1.z + v.w * w1.w;
    float b = v.x * w2.x + v.y * w2.y + v.z * w2.z + v.w * w2.w;
#pragma unroll
    for (int off = 16; off > 0; off >>= 1) {
        a += __shfl_xor(a, off, 32);
        b += __shfl_xor(b, off, 32);
    }
    if (lane == 0) { s1[wid] = a; s2[wid] = b; }
}

// e = leaky_relu(s1[src,rel] + s2[dst,rel] + ba); segment max into segmax[dst].
__global__ void edge_logit_kernel(const int* __restrict__ src,
                                  const int* __restrict__ dst,
                                  const int* __restrict__ rel,
                                  const float* __restrict__ s1,
                                  const float* __restrict__ s2,
                                  const float* __restrict__ ba,
                                  float* __restrict__ e,
                                  float* __restrict__ segmax)
{
    int i = blockIdx.x * blockDim.x + threadIdx.x;
    if (i >= NEDGES) return;
    int s = src[i], d = dst[i], r = rel[i];
    float v = s1[s * NREL + r] + s2[d * NREL + r] + ba[0];
    v = (v > 0.0f) ? v : 0.2f * v;
    e[i] = v;
    atomicMaxFloat(&segmax[d], v);
}

// ex = exp(e - max[dst]); denom[dst] += ex   (ex overwrites e)
__global__ void edge_exp_kernel(const int* __restrict__ dst,
                                const float* __restrict__ segmax,
                                float* __restrict__ e,
                                float* __restrict__ denom)
{
    int i = blockIdx.x * blockDim.x + threadIdx.x;
    if (i >= NEDGES) return;
    int d = dst[i];
    float ex = __expf(e[i] - segmax[d]);
    e[i] = ex;
    atomicAdd(&denom[d], ex);
}

// agg[dst,:] += (ex/denom[dst]) * Ht[src,rel,:]. One wave per edge, float4/lane.
__global__ void edge_agg_kernel(const int* __restrict__ src,
                                const int* __restrict__ dst,
                                const int* __restrict__ rel,
                                const float* __restrict__ ex,
                                const float* __restrict__ denom,
                                const float* __restrict__ Ht,
                                float* __restrict__ agg)
{
    const int wid  = (int)(((long long)blockIdx.x * blockDim.x + threadIdx.x) >> 5);
    const int lane = threadIdx.x & 31;
    if (wid >= NEDGES) return;
    int s = src[wid], d = dst[wid], r = rel[wid];
    float alpha = ex[wid] / denom[d];
    float4 v = ((const float4*)(Ht + ((long long)s * NREL + r) * FOUT))[lane];
    float* ag = agg + (long long)d * FOUT + lane * 4;
    atomicAdd(ag + 0, alpha * v.x);
    atomicAdd(ag + 1, alpha * v.y);
    atomicAdd(ag + 2, alpha * v.z);
    atomicAdd(ag + 3, alpha * v.w);
}

// h_new = agg + h @ loop_weight (in place over agg) + bf16 copy for next GEMM.
__global__ void hnew_gemm_kernel(const unsigned short* __restrict__ hB,
                                 const unsigned short* __restrict__ loopT,
                                 float* __restrict__ agg,
                                 unsigned short* __restrict__ hnewB)
{
    const int lane = threadIdx.x & 31;
    const int wave = threadIdx.x >> 5;
    const int half = lane >> 4;
    const int mr   = lane & 15;
    const int tile_m = blockIdx.x;

    const unsigned short* Arow = hB + (long long)(tile_m * 16 + mr) * FIN;
    const unsigned short* Brow = loopT + (long long)(wave * 16 + mr) * FIN;
    const int col = wave * 16 + mr;

    v8f acc;
#pragma unroll
    for (int v = 0; v < 8; ++v) {
        int row = tile_m * 16 + v + half * 8;
        acc[v] = agg[(long long)row * FOUT + col];
    }
    wmma_k_loop(Arow, Brow, FIN, half, acc);
#pragma unroll
    for (int v = 0; v < 8; ++v) {
        int row = tile_m * 16 + v + half * 8;
        agg[(long long)row * FOUT + col]   = acc[v];
        hnewB[(long long)row * FOUT + col] = f2bf(acc[v]);
    }
}

// sum_nodes[col] += sum over a 256-row chunk of h_new.
__global__ void colsum_kernel(const float* __restrict__ hnew,
                              float* __restrict__ sum_nodes)
{
    int col = threadIdx.x;                 // blockDim = 128
    int r0  = blockIdx.x * 256;
    int r1  = r0 + 256; if (r1 > NNODES) r1 = NNODES;
    float s = 0.0f;
    for (int r = r0; r < r1; ++r) s += hnew[(long long)r * FOUT + col];
    atomicAdd(&sum_nodes[col], s);
}

// n_out = relu(h_new @ Wn + bn)
__global__ void nout_gemm_kernel(const unsigned short* __restrict__ hnewB,
                                 const unsigned short* __restrict__ WnT,
                                 const float* __restrict__ bn,
                                 float* __restrict__ nout)
{
    const int lane = threadIdx.x & 31;
    const int wave = threadIdx.x >> 5;
    const int half = lane >> 4;
    const int mr   = lane & 15;
    const int tile_m = blockIdx.x;

    const unsigned short* Arow = hnewB + (long long)(tile_m * 16 + mr) * FIN;
    const unsigned short* Brow = WnT + (long long)(wave * 16 + mr) * FIN;
    const int col = wave * 16 + mr;

    v8f acc = {};
    wmma_k_loop(Arow, Brow, FIN, half, acc);
    float bias = bn[col];
#pragma unroll
    for (int v = 0; v < 8; ++v) {
        int row = tile_m * 16 + v + half * 8;
        float x = acc[v] + bias;
        nout[(long long)row * FOUT + col] = (x > 0.0f) ? x : 0.0f;
    }
}

// u_new = relu([u, sum_nodes] @ Wg + bg)  -- tiny, one block.
__global__ void unew_kernel(const float* __restrict__ u,
                            const float* __restrict__ sum_nodes,
                            const float* __restrict__ Wg,
                            const float* __restrict__ bg,
                            float* __restrict__ out)
{
    int t = threadIdx.x;                   // blockDim = 256
    float acc = bg[t];
    for (int j = 0; j < NGLOB; ++j)  acc += u[j] * Wg[j * NGLOB + t];
    for (int j = 0; j < FOUT;  ++j)  acc += sum_nodes[j] * Wg[(NGLOB + j) * NGLOB + t];
    out[t] = (acc > 0.0f) ? acc : 0.0f;
}

// ---------- launcher ----------

extern "C" void kernel_launch(void* const* d_in, const int* in_sizes, int n_in,
                              void* d_out, int out_size, void* d_ws, size_t ws_size,
                              hipStream_t stream)
{
    const float* h      = (const float*)d_in[0];
    const float* u      = (const float*)d_in[1];
    const float* weight = (const float*)d_in[2];
    const float* loopw  = (const float*)d_in[3];
    const float* Wa     = (const float*)d_in[4];
    const float* ba     = (const float*)d_in[5];
    const float* Wn     = (const float*)d_in[6];
    const float* bn     = (const float*)d_in[7];
    const float* Wg     = (const float*)d_in[8];
    const float* bg     = (const float*)d_in[9];
    const int*   src    = (const int*)d_in[10];
    const int*   dst    = (const int*)d_in[11];
    const int*   rel    = (const int*)d_in[12];
    float* out = (float*)d_out;            // [0..255]=u_new, then n_out (N*FOUT)

    char* w = (char*)d_ws;
    auto alloc = [&](size_t bytes) {
        char* p = w;
        w += (bytes + 255) & ~(size_t)255;
        return p;
    };
    float*          Ht      = (float*)alloc(sizeof(float) * (size_t)NNODES * NREL * FOUT);
    unsigned short* hB      = (unsigned short*)alloc(sizeof(unsigned short) * (size_t)NNODES * FIN);
    unsigned short* wT      = (unsigned short*)alloc(sizeof(unsigned short) * (size_t)NREL * FIN * FOUT);
    unsigned short* loopT   = (unsigned short*)alloc(sizeof(unsigned short) * FIN * FOUT);
    unsigned short* WnT     = (unsigned short*)alloc(sizeof(unsigned short) * FIN * FOUT);
    unsigned short* hnewB   = (unsigned short*)alloc(sizeof(unsigned short) * (size_t)NNODES * FOUT);
    float*          s1      = (float*)alloc(sizeof(float) * (size_t)NNODES * NREL);
    float*          s2      = (float*)alloc(sizeof(float) * (size_t)NNODES * NREL);
    float*          ebuf    = (float*)alloc(sizeof(float) * (size_t)NEDGES);
    float*          segmax  = (float*)alloc(sizeof(float) * (size_t)NNODES);
    float*          denom   = (float*)alloc(sizeof(float) * (size_t)NNODES);
    float*          agg     = (float*)alloc(sizeof(float) * (size_t)NNODES * FOUT);
    float*          sumn    = (float*)alloc(sizeof(float) * FOUT);

    const int TM = NNODES / 16;            // 3125 (exact)

    prep_kernel<<<(NNODES * FIN + 255) / 256, 256, 0, stream>>>(
        h, weight, loopw, Wn, hB, wT, loopT, WnT, segmax, denom, agg, sumn);

    ht_gemm_kernel<<<dim3(TM, NREL), 256, 0, stream>>>(hB, wT, Ht);

    score_kernel<<<(NNODES * NREL * 32 + 255) / 256, 256, 0, stream>>>(Ht, Wa, s1, s2);

    edge_logit_kernel<<<(NEDGES + 255) / 256, 256, 0, stream>>>(
        src, dst, rel, s1, s2, ba, ebuf, segmax);

    edge_exp_kernel<<<(NEDGES + 255) / 256, 256, 0, stream>>>(dst, segmax, ebuf, denom);

    edge_agg_kernel<<<(NEDGES * 32 + 255) / 256, 256, 0, stream>>>(
        src, dst, rel, ebuf, denom, Ht, agg);

    hnew_gemm_kernel<<<dim3(TM), 256, 0, stream>>>(hB, loopT, agg, hnewB);

    colsum_kernel<<<(NNODES + 255) / 256, 128, 0, stream>>>(agg, sumn);

    nout_gemm_kernel<<<dim3(TM), 256, 0, stream>>>(hnewB, WnT, bn, out + NGLOB);

    unew_kernel<<<1, 256, 0, stream>>>(u, sumn, Wg, bg, out);
}
